// BooleanReservoir_67267777790486
// MI455X (gfx1250) — compile-verified, matrix operand order checked by reading, not run on hardware
//
#include <hip/hip_runtime.h>
#include <hip/hip_bf16.h>

typedef unsigned long long u64;
typedef __attribute__((ext_vector_type(16))) _Float16 v16h;
typedef __attribute__((ext_vector_type(8)))  _Float16 v8h;
typedef __attribute__((ext_vector_type(8)))  float    v8f;

#define N_NODES 50000
#define N_PAD   50016   // next multiple of 32 -> uniform WMMA K loop, zero-padded
#define K_MAX   10
#define BATCH   64
#define T_STEPS 32
#define INPUT_BITS 64
#define LUT_WORDS 16    // 1024 bits / 64
#define CHUNKS  (N_PAD / 32)                    // 1563 K-chunks of 32
#define KSPLIT  16                              // K-parallel blocks per tile
#define CPS     ((CHUNKS + KSPLIT - 1) / KSPLIT) // 98 chunks per split

// ---------------- packing kernels ----------------

__global__ void pack_init_kernel(const int* __restrict__ init, u64* __restrict__ s, int n_nodes) {
    int n = blockIdx.x * blockDim.x + threadIdx.x;
    if (n < n_nodes) s[n] = (init[n] & 1) ? ~0ull : 0ull;
}

// one thread packs 64 consecutive lut ints into one u64 word
__global__ void pack_lut_kernel(const int* __restrict__ lut, u64* __restrict__ lutp, int total_words) {
    int t = blockIdx.x * blockDim.x + threadIdx.x;
    if (t >= total_words) return;
    int base = t * 64;              // t = n*16 + w  ->  flat lut idx n*1024 + w*64
    u64 bits = 0;
    #pragma unroll
    for (int j = 0; j < 64; ++j) bits |= (u64)(lut[base + j] & 1) << j;
    lutp[t] = bits;
}

// x is (B=64, T=32, 64) int32; pack across batch: xp[t*64+i] bit b = x[b][t][i]
__global__ void pack_x_kernel(const int* __restrict__ x, u64* __restrict__ xp) {
    int t = blockIdx.x * blockDim.x + threadIdx.x;   // 0 .. 2047
    if (t >= T_STEPS * INPUT_BITS) return;
    u64 bits = 0;
    for (int b = 0; b < BATCH; ++b)
        bits |= (u64)(x[b * (T_STEPS * INPUT_BITS) + t] & 1) << b;
    xp[t] = bits;
}

// ---------------- per-timestep kernels ----------------

// xor-inject the packed input bits into the current state (set-scatter semantics:
// last duplicate occurrence wins, deterministically)
__global__ void inject_kernel(u64* __restrict__ s, const u64* __restrict__ xp,
                              const int* __restrict__ input_nodes, int t) {
    int i = threadIdx.x;                 // 64 threads, 1 block
    int node = input_nodes[i];
    bool win = true;
    for (int j = i + 1; j < INPUT_BITS; ++j)
        if (input_nodes[j] == node) { win = false; break; }
    if (win) s[node] = s[node] ^ xp[t * INPUT_BITS + i];
}

// one thread per node: gather 10 packed neighbor words, evaluate 10-input LUT
// bit-parallel across the 64 batches. LUT row staged in LDS for dynamic indexing.
__global__ __launch_bounds__(256) void update_kernel(
        const u64* __restrict__ s_cur, u64* __restrict__ s_next,
        const int* __restrict__ adj, const int* __restrict__ mask,
        const u64* __restrict__ lutp, int n_nodes) {
    __shared__ u64 lutsh[LUT_WORDS * 256];
    const int tid = threadIdx.x;
    const int n = blockIdx.x * blockDim.x + tid;
    if (n >= n_nodes) return;

    u64 w[K_MAX];
    #pragma unroll
    for (int k = 0; k < K_MAX; ++k) {
        int a = adj[n * K_MAX + k];
        int m = mask[n * K_MAX + k];
        w[k] = m ? s_cur[a] : 0ull;
    }
    #pragma unroll
    for (int j = 0; j < LUT_WORDS; ++j)
        lutsh[j * 256 + tid] = lutp[(size_t)n * LUT_WORDS + j];   // bank-friendly layout

    u64 res = 0;
    #pragma unroll 4
    for (int b = 0; b < BATCH; ++b) {
        unsigned idx = 0;
        #pragma unroll
        for (int k = 0; k < K_MAX; ++k)          // adj[0] -> MSB (pow2 = 512 >> k)
            idx = (idx << 1) | (unsigned)((w[k] >> b) & 1ull);
        u64 lw = lutsh[(idx >> 6) * 256 + tid];
        res |= ((lw >> (idx & 63)) & 1ull) << b;
    }
    s_next[n] = res;
}

// ---------------- GEMM prep: fragment-layout f16 operands, zero-padded ----------------

// S16[b][n] = (f16) state bit(b, n); batch-major rows of stride N_PAD (16B aligned)
__global__ void expand_states_kernel(const u64* __restrict__ s, _Float16* __restrict__ S16) {
    int n = blockIdx.x * blockDim.x + threadIdx.x;
    int b = blockIdx.y;
    if (n >= N_PAD) return;
    _Float16 v = (_Float16)0.0f;
    if (n < N_NODES) v = (_Float16)(float)((s[n] >> b) & 1ull);
    S16[(size_t)b * N_PAD + n] = v;
}

// Wh[i][n] = (f16) W[i][n]; rows of stride N_PAD, zero tail
__global__ void convert_w_kernel(const float* __restrict__ W, _Float16* __restrict__ Wh) {
    int n = blockIdx.x * blockDim.x + threadIdx.x;
    int i = blockIdx.y;                   // 0..15
    if (n >= N_PAD) return;
    _Float16 v = (_Float16)0.0f;
    if (n < N_NODES) v = (_Float16)W[i * N_NODES + n];
    Wh[(size_t)i * N_PAD + n] = v;
}

// ---------------- readout GEMM: out(64x16) = states(64x50000) * W^T + b ----------------
// grid (4 tiles x KSPLIT K-splits), one wave each. Software-pipelined:
// loads of chunk c+1 are in flight while v_wmma consumes chunk c (no full
// loadcnt drain before the WMMA). EXEC stays all-1s throughout.
__global__ __launch_bounds__(32) void gemm_wmma_kernel(
        const _Float16* __restrict__ S16, const _Float16* __restrict__ Wh,
        float* __restrict__ partial) {
    const int lane  = threadIdx.x;      // 0..31
    const int tile  = blockIdx.x;       // 0..3 -> batch rows [tile*16, tile*16+16)
    const int split = blockIdx.y;       // 0..KSPLIT-1
    const int half  = lane >> 4;        // K-half selector per ISA 16-bit A/B layout
    const int mn    = lane & 15;        // M row (A) / N col (B)

    const int c0 = split * CPS;
    const int c1 = (c0 + CPS < CHUNKS) ? (c0 + CPS) : CHUNKS;
    const int nchunks = c1 - c0;        // uniform per block, always >= 1

    const _Float16* aptr = S16 + (size_t)(tile * 16 + mn) * N_PAD + half * 8 + (size_t)c0 * 32;
    const _Float16* bptr = Wh  + (size_t)mn * N_PAD + half * 8 + (size_t)c0 * 32;

    v8f acc = {};
    v8h alo = *(const v8h*)(aptr);
    v8h ahi = *(const v8h*)(aptr + 16);
    v8h blo = *(const v8h*)(bptr);
    v8h bhi = *(const v8h*)(bptr + 16);
    for (int c = 1; c < nchunks; ++c) {
        v8h alo2 = *(const v8h*)(aptr + c * 32);
        v8h ahi2 = *(const v8h*)(aptr + c * 32 + 16);
        v8h blo2 = *(const v8h*)(bptr + c * 32);
        v8h bhi2 = *(const v8h*)(bptr + c * 32 + 16);
        v16h a, b;
        #pragma unroll
        for (int e = 0; e < 8; ++e) {
            a[e] = alo[e]; a[e + 8] = ahi[e];
            b[e] = blo[e]; b[e + 8] = bhi[e];
        }
        acc = __builtin_amdgcn_wmma_f32_16x16x32_f16(
            false, a, false, b, (short)0, acc, false, false);
        alo = alo2; ahi = ahi2; blo = blo2; bhi = bhi2;
    }
    {   // epilogue chunk
        v16h a, b;
        #pragma unroll
        for (int e = 0; e < 8; ++e) {
            a[e] = alo[e]; a[e + 8] = ahi[e];
            b[e] = blo[e]; b[e + 8] = bhi[e];
        }
        acc = __builtin_amdgcn_wmma_f32_16x16x32_f16(
            false, a, false, b, (short)0, acc, false, false);
    }
    // store 16x16 f32 partial tile for this (tile, split)
    float* p = partial + ((size_t)tile * KSPLIT + split) * 256;
    #pragma unroll
    for (int r = 0; r < 8; ++r) {
        int M = r + half * 8;                     // C/D layout: lanes16-31 hold M+8
        p[M * 16 + mn] = acc[r];
    }
}

// fixed-order (deterministic) reduction over K-splits, plus bias
__global__ void gemm_reduce_kernel(const float* __restrict__ partial,
                                   const float* __restrict__ bias,
                                   float* __restrict__ out) {
    int e = blockIdx.x * blockDim.x + threadIdx.x;   // 0..1023
    if (e >= BATCH * 16) return;
    int row = e >> 4, col = e & 15;
    int tile = row >> 4, m = row & 15;
    float acc = bias[col];
    #pragma unroll
    for (int s = 0; s < KSPLIT; ++s)
        acc += partial[((size_t)tile * KSPLIT + s) * 256 + m * 16 + col];
    out[e] = acc;
}

// ---------------- host launcher ----------------

extern "C" void kernel_launch(void* const* d_in, const int* in_sizes, int n_in,
                              void* d_out, int out_size, void* d_ws, size_t ws_size,
                              hipStream_t stream) {
    const int* x           = (const int*)d_in[0];
    const int* adj_list    = (const int*)d_in[1];
    const int* adj_mask    = (const int*)d_in[2];
    const int* lut         = (const int*)d_in[3];
    const int* init_states = (const int*)d_in[4];
    const int* input_nodes = (const int*)d_in[5];
    const float* W         = (const float*)d_in[6];
    const float* b         = (const float*)d_in[7];
    float* out             = (float*)d_out;

    char* ws = (char*)d_ws;
    u64*      s0      = (u64*)(ws + 0);                      //   400,000 B
    u64*      s1      = (u64*)(ws + (512u * 1024u));         //   400,000 B
    u64*      lutp    = (u64*)(ws + (1024u * 1024u));        // 6,400,000 B
    u64*      xp      = (u64*)(ws + (8u * 1024u * 1024u));   //    16,384 B
    _Float16* S16     = (_Float16*)(ws + (8u*1024u*1024u + 128u*1024u)); // 6,402,048 B
    _Float16* Wh      = (_Float16*)(ws + (15u * 1024u * 1024u));         // 1,600,512 B
    float*    partial = (float*)(ws + (17u * 1024u * 1024u));            //    65,536 B

    // pack phase
    pack_init_kernel<<<(N_NODES + 255) / 256, 256, 0, stream>>>(init_states, s0, N_NODES);
    pack_lut_kernel<<<(N_NODES * LUT_WORDS + 255) / 256, 256, 0, stream>>>(
        lut, lutp, N_NODES * LUT_WORDS);
    pack_x_kernel<<<(T_STEPS * INPUT_BITS + 255) / 256, 256, 0, stream>>>(x, xp);
    // W -> f16 fragment rows (independent of the recurrence; do it up front)
    convert_w_kernel<<<dim3((N_PAD + 255) / 256, 16), 256, 0, stream>>>(W, Wh);

    // 32 serialized reservoir steps (stream order = global sync)
    u64* cur = s0;
    u64* nxt = s1;
    for (int t = 0; t < T_STEPS; ++t) {
        inject_kernel<<<1, INPUT_BITS, 0, stream>>>(cur, xp, input_nodes, t);
        update_kernel<<<(N_NODES + 255) / 256, 256, 0, stream>>>(
            cur, nxt, adj_list, adj_mask, lutp, N_NODES);
        u64* tmp = cur; cur = nxt; nxt = tmp;
    }
    // T even -> final state back in s0 (== cur)

    // expand final packed states into WMMA A-fragment rows, then readout GEMM
    expand_states_kernel<<<dim3((N_PAD + 255) / 256, BATCH), 256, 0, stream>>>(cur, S16);
    gemm_wmma_kernel<<<dim3(4, KSPLIT), 32, 0, stream>>>(S16, Wh, partial);
    gemm_reduce_kernel<<<(BATCH * 16 + 255) / 256, 256, 0, stream>>>(partial, b, out);

    (void)in_sizes; (void)n_in; (void)out_size; (void)ws_size;
}
